// AdaptiveConvND_77043123355920
// MI455X (gfx1250) — matrix operand order repaired and checked by reading, not compiled, not run on hardware
//
#include <hip/hip_runtime.h>
#include <hip/hip_bf16.h>
#include <stdint.h>

// ---------------- problem constants (from reference) ----------------
#define BB    2
#define LL    4096
#define CC    1024
#define HH    8
#define POSN  16
#define DD    128          // CC / HH
#define SSAMP 33           // 2*HALF_S + 1
#define MROWS (BB*LL)      // 8192

// ---------------- WMMA types ----------------
typedef __attribute__((ext_vector_type(16))) __bf16 v16bf;
typedef __attribute__((ext_vector_type(8)))  float  v8f;

union FragU { uint4 q[2]; v16bf v; };

// ---------------- small device helpers ----------------
__device__ __forceinline__ float siluf(float v) { return v / (1.f + expf(-v)); }
__device__ __forceinline__ float sigf(float v)  { return 1.f / (1.f + expf(-v)); }

__device__ __forceinline__ float wredsum(float v) {
#pragma unroll
  for (int m = 16; m >= 1; m >>= 1) v += __shfl_xor(v, m, 32);
  return v;
}
__device__ __forceinline__ float wredmax(float v) {
#pragma unroll
  for (int m = 16; m >= 1; m >>= 1) v = fmaxf(v, __shfl_xor(v, m, 32));
  return v;
}

__device__ __forceinline__ void store_res(float* p, float v) { *p = v; }
__device__ __forceinline__ void store_res(__hip_bfloat16* p, float v) { *p = __float2bfloat16(v); }

// ---- CDNA5 async global->LDS copy (ASYNCcnt path, ISA 08_async_tensor) ----
// global_load_async_to_lds_b128: VDST = per-lane LDS byte address, VADDR = per-lane
// 64-bit global address, SADDR = NULL ("off").  Generic LDS pointers carry the LDS
// byte offset in addr[31:0], so truncating a __shared__ pointer gives VDST.
__device__ __forceinline__ void async_ld16(void* lds_ptr, const void* gptr) {
  asm volatile("global_load_async_to_lds_b128 %0, %1, off"
               :: "v"((uint32_t)(uintptr_t)lds_ptr),
                  "v"((unsigned long long)(uintptr_t)gptr)
               : "memory");
}
__device__ __forceinline__ void wait_async0() {
  asm volatile("s_wait_asynccnt 0x0" ::: "memory");
}

// ---------------- conversion / transpose prep ----------------
__global__ void cvt_kernel(const float* __restrict__ in, __hip_bfloat16* __restrict__ out, size_t n) {
  size_t i = (size_t)blockIdx.x * 256 + threadIdx.x;
  if (i < n) out[i] = __float2bfloat16(in[i]);
}

// in: f32 row-major (K,N)  ->  out: bf16 row-major (N,K)
__global__ void transpose_cvt(const float* __restrict__ in, __hip_bfloat16* __restrict__ out, int K, int N) {
  int idx = blockIdx.x * 256 + threadIdx.x;
  if (idx < K * N) {
    int k = idx / N, n = idx % N;
    out[(size_t)n * K + k] = __float2bfloat16(in[idx]);
  }
}

// ---------------- wave params: freq/decay per (b,l,h); sample idx + valid per (b,l,s) ----------------
__global__ void wave_params_kernel(const float* __restrict__ x, const float* __restrict__ wave_w,
                                   const float* __restrict__ wave_b,
                                   float* __restrict__ freq, float* __restrict__ decay,
                                   int* __restrict__ sidx, float* __restrict__ vmask) {
  __shared__ float xs[CC];
  __shared__ float wv[24];
  __shared__ float fph[16];   // [0..7]=freq, [8..15]=phase
  __shared__ float avg[2];

  const int g    = blockIdx.x;           // g = b*L + l
  const int l    = g % LL;
  const int tid  = threadIdx.x;
  const int wave = tid >> 5;
  const int lane = tid & 31;

  for (int i = tid; i < CC; i += 256) xs[i] = x[(size_t)g * CC + i];
  __syncthreads();

  // 24 dot products of length 1024; wave w owns outputs {w, 8+w, 16+w}
#pragma unroll
  for (int jj = 0; jj < 3; ++jj) {
    int j = jj * 8 + wave;
    float p = 0.f;
    for (int k = lane; k < CC; k += 32) p += xs[k] * wave_w[k * 24 + j];
    p = wredsum(p);
    if (lane == 0) wv[j] = siluf(p + wave_b[j]);
  }
  __syncthreads();

  if (tid < 8) {                                   // freq
    float fr = 1.f + 15.f * sigf(wv[tid]);
    freq[(size_t)g * HH + tid] = fr;
    fph[tid] = fr;
  } else if (tid < 16) {                           // phase
    int h = tid - 8;
    fph[8 + h] = tanhf(wv[8 + h]) * 16.f;
  } else if (tid < 24) {                           // decay
    int h = tid - 16;
    decay[(size_t)g * HH + h] = 0.5f + 9.5f * sigf(wv[16 + h]);
  }
  __syncthreads();

  if (tid == 0) {
    float fa = 0.f, pa = 0.f;
#pragma unroll
    for (int h = 0; h < 8; ++h) { fa += fph[h]; pa += fph[8 + h]; }
    avg[0] = fa * 0.125f;
    avg[1] = pa * 0.125f;
  }
  __syncthreads();

  if (tid < SSAMP) {
    float pos = (float)l + (float)(tid - 16) * avg[0] + avg[1];
    bool  ok  = (pos >= 0.f) && (pos < (float)LL);
    int   ip  = (int)pos;                // truncation like astype(int32)
    ip = min(max(ip, 0), LL - 1);
    sidx[(size_t)g * SSAMP + tid]  = ip;
    vmask[(size_t)g * SSAMP + tid] = ok ? 1.f : 0.f;
  }
}

// ---------------- adaptive softmax-gather conv: one wave per (b,l,h) ----------------
__global__ void attend_kernel(const float* __restrict__ x, const float* __restrict__ queries,
                              const float* __restrict__ freq, const float* __restrict__ decay,
                              const int* __restrict__ sidx, const float* __restrict__ vmask,
                              const float* __restrict__ key_w,
                              float* __restrict__ outf, __hip_bfloat16* __restrict__ outbf) {
  __shared__ float kws[POSN];
  const int g    = blockIdx.x;     // b*L + l
  const int h    = threadIdx.x >> 5;
  const int lane = threadIdx.x & 31;
  const int b    = g / LL;

  if (threadIdx.x < POSN) kws[threadIdx.x] = key_w[threadIdx.x];
  __syncthreads();

  const float q  = (lane < POSN) ? queries[(size_t)g * (HH * POSN) + h * POSN + lane] : 0.f;
  const float f  = freq[(size_t)g * HH + h];
  const float dc = fmaxf(decay[(size_t)g * HH + h], 0.1f);

  // per-tap logits/env/idx (executed by all lanes so shuffles stay convergent)
  auto calc = [&](int s, float& logit, float& env, int& ii, float& vm) {
    float rel = fabsf((float)(s - 16)) * f;
    float lg = 0.f;
#pragma unroll
    for (int p = 0; p < POSN; ++p) {
      float kv = rel * kws[p];
      lg += __shfl(q, p, 32) * siluf(kv);
    }
    logit = lg * 0.25f;                 // SCALE = POS^-0.5
    env   = expf(-rel / dc);
    ii    = sidx[(size_t)g * SSAMP + s];
    vm    = vmask[(size_t)g * SSAMP + s];
  };

  float lg0, env0, vm0; int i0;
  calc(lane, lg0, env0, i0, vm0);       // taps 0..31 (one per lane)
  float lg1, env1, vm1; int i1;
  calc(32, lg1, env1, i1, vm1);         // tap 32 (uniform across lanes)

  float l0m = (vm0 > 0.f) ? lg0 : -1e30f;
  float l1m = (vm1 > 0.f) ? lg1 : -1e30f;
  float mx  = wredmax(fmaxf(l0m, l1m));

  float e0 = (vm0 > 0.f) ? expf(lg0 - mx) : 0.f;
  float e1 = (vm1 > 0.f) ? expf(lg1 - mx) : 0.f;
  float ssum = wredsum(e0) + e1;        // e1 uniform -> add once after reduction
  ssum = fmaxf(ssum, 1e-30f);

  float w0 = (e0 / ssum) * env0 * vm0;
  float w1 = (e1 / ssum) * env1 * vm1;
  float den = wredsum(w0) + w1 + 1e-8f;
  w0 /= den;  w1 /= den;

  float ax = 0.f, ay = 0.f, az = 0.f, aw = 0.f;
  const size_t xrow0 = (size_t)b * LL;
  const int    coff  = h * DD + lane * 4;
#pragma unroll 1
  for (int s = 0; s < SSAMP; ++s) {
    float ws; int ip;
    if (s < 32) { ws = __shfl(w0, s, 32); ip = __shfl(i0, s, 32); }
    else        { ws = w1;                ip = i1; }
    if (ws != 0.f) {                       // wave-uniform branch
      const float4 v = *(const float4*)(x + (xrow0 + (size_t)ip) * CC + coff);
      ax += ws * v.x; ay += ws * v.y; az += ws * v.z; aw += ws * v.w;
    }
  }

  const size_t o = (size_t)g * CC + coff;
  outf[o + 0] = ax; outf[o + 1] = ay; outf[o + 2] = az; outf[o + 3] = aw;
  outbf[o + 0] = __float2bfloat16(ax);
  outbf[o + 1] = __float2bfloat16(ay);
  outbf[o + 2] = __float2bfloat16(az);
  outbf[o + 3] = __float2bfloat16(aw);
}

// ---------------- bf16 WMMA GEMM with async-LDS double buffering ----------------
// C = act(A[M,K] @ Bt^T + bias), Bt pre-transposed to (N,K) bf16.
// Block: 256 thr = 8 waves (4 row-groups x 2 col-groups); block tile 128x128;
// each wave computes 32x64 (8 WMMA accumulators).  K consumed 32 at a time.
// LDS rows padded to 40 bf16 (80B): ds_load_b128 fragment reads are
// bank-conflict-free (stride 20 banks covers all 64 banks exactly once).
// ACT 0: silu(v).  ACT 1: sigmoid(v) * mul[m,n]  (SE excitation fused).
template <int ACT, typename OutT>
__global__ void __launch_bounds__(256)
gemm_bf16_wmma(const __hip_bfloat16* __restrict__ A, const __hip_bfloat16* __restrict__ Bt,
               const float* __restrict__ bias, const float* __restrict__ mul,
               OutT* __restrict__ C, int M, int N, int K) {
  __shared__ __align__(16) __hip_bfloat16 A_s[2][128][40];
  __shared__ __align__(16) __hip_bfloat16 B_s[2][128][40];

  const int tid  = threadIdx.x;
  const int wv   = tid >> 5;
  const int lane = tid & 31;
  const int half = lane >> 4;
  const int l16  = lane & 15;
  const int wr   = wv >> 1;                 // 0..3 : 32-row group
  const int wc   = wv & 1;                  // 0..1 : 64-col group
  const int rowBase = blockIdx.y * 128;
  const int colBase = blockIdx.x * 128;

  // stage one 128x32 A tile + 128x32 B tile into LDS buffer `buf` (async).
  // 512 x 16B transfers per tile; 256 threads -> 2 per tile per thread.
  auto stage = [&](int buf, int k0) {
#pragma unroll
    for (int j = 0; j < 2; ++j) {
      int i = tid + j * 256;
      int row = i >> 2, seg = i & 3;
      async_ld16(&A_s[buf][row][seg * 8],
                 A + (size_t)(rowBase + row) * K + k0 + seg * 8);
      async_ld16(&B_s[buf][row][seg * 8],
                 Bt + (size_t)(colBase + row) * K + k0 + seg * 8);
    }
  };

  // ISA 7.12.2 fragment reads out of LDS (80B padded rows -> 5 uint4 per row)
  auto fragA = [&](int buf, int row) -> v16bf {
    FragU f;
    const uint4* p = (const uint4*)&A_s[buf][row][0];
    f.q[0] = p[half];          // k: half*8 .. +7
    f.q[1] = p[2 + half];      // k: 16+half*8 .. +7
    return f.v;
  };
  auto fragB = [&](int buf, int col) -> v16bf {
    FragU f;
    const uint4* p = (const uint4*)&B_s[buf][col][0];
    f.q[0] = p[half];          // k: half*8 .. +7
    f.q[1] = p[2 + half];      // k: 16+half*8 .. +7
    return f.v;
  };

  v8f acc[2][4] = {};

  auto compute = [&](int buf) {
    v16bf a0 = fragA(buf, wr * 32 + l16);
    v16bf a1 = fragA(buf, wr * 32 + 16 + l16);
#pragma unroll
    for (int ct = 0; ct < 4; ++ct) {
      v16bf b = fragB(buf, wc * 64 + ct * 16 + l16);
      acc[0][ct] = __builtin_amdgcn_wmma_f32_16x16x32_bf16(
          false, a0, false, b, (short)0, acc[0][ct], false, false);
      acc[1][ct] = __builtin_amdgcn_wmma_f32_16x16x32_bf16(
          false, a1, false, b, (short)0, acc[1][ct], false, false);
    }
  };

  stage(0, 0);
  wait_async0();
  __syncthreads();

  int cur = 0;
  for (int k0 = 32; k0 < K; k0 += 32) {
    stage(cur ^ 1, k0);      // async prefetch next K-step while computing
    compute(cur);
    wait_async0();           // my async copies landed in LDS
    __syncthreads();         // everyone's copies landed / everyone done reading
    cur ^= 1;
  }
  compute(cur);

  // epilogue: acc layout (ISA): VGPR r <-> m = r + 8*half, lane&15 <-> n
#pragma unroll
  for (int rt = 0; rt < 2; ++rt) {
#pragma unroll
    for (int ct = 0; ct < 4; ++ct) {
      int n = colBase + wc * 64 + ct * 16 + l16;
      float bv = bias ? bias[n] : 0.f;
#pragma unroll
      for (int r = 0; r < 8; ++r) {
        int m = rowBase + wr * 32 + rt * 16 + r + half * 8;
        float v = acc[rt][ct][r] + bv;
        float res;
        if (ACT == 0) res = siluf(v);
        else          res = sigf(v) * mul[(size_t)m * N + n];
        store_res(C + (size_t)m * N + n, res);
      }
    }
  }
}

// ---------------- launcher ----------------
extern "C" void kernel_launch(void* const* d_in, const int* in_sizes, int n_in,
                              void* d_out, int out_size, void* d_ws, size_t ws_size,
                              hipStream_t stream) {
  (void)in_sizes; (void)n_in; (void)out_size; (void)ws_size;

  const float* x       = (const float*)d_in[0];
  const float* wave_w  = (const float*)d_in[1];
  const float* wave_b  = (const float*)d_in[2];
  const float* query_w = (const float*)d_in[3];
  const float* query_b = (const float*)d_in[4];
  const float* key_w   = (const float*)d_in[5];
  const float* out_w   = (const float*)d_in[6];
  const float* se1_w   = (const float*)d_in[7];
  const float* se1_b   = (const float*)d_in[8];
  const float* se2_w   = (const float*)d_in[9];
  const float* se2_b   = (const float*)d_in[10];

  char* ws = (char*)d_ws;
  size_t off = 0;
  auto alloc = [&](size_t bytes) -> char* {
    char* p = ws + off;
    off += (bytes + 255) & ~(size_t)255;
    return p;
  };

  __hip_bfloat16* x_bf  = (__hip_bfloat16*)alloc((size_t)MROWS * CC * 2);
  __hip_bfloat16* qwT   = (__hip_bfloat16*)alloc((size_t)128 * CC * 2);
  __hip_bfloat16* se1T  = (__hip_bfloat16*)alloc((size_t)256 * CC * 2);
  __hip_bfloat16* se2T  = (__hip_bfloat16*)alloc((size_t)CC * 256 * 2);
  __hip_bfloat16* owT   = (__hip_bfloat16*)alloc((size_t)CC * CC * 2);
  float* queries        = (float*)alloc((size_t)MROWS * 128 * 4);
  float* freq           = (float*)alloc((size_t)MROWS * HH * 4);
  float* decay          = (float*)alloc((size_t)MROWS * HH * 4);
  int*   sidx           = (int*)alloc((size_t)MROWS * SSAMP * 4);
  float* vmask          = (float*)alloc((size_t)MROWS * SSAMP * 4);
  float* outf           = (float*)alloc((size_t)MROWS * CC * 4);
  __hip_bfloat16* outbf = (__hip_bfloat16*)alloc((size_t)MROWS * CC * 2);
  __hip_bfloat16* t_bf  = (__hip_bfloat16*)alloc((size_t)MROWS * 256 * 2);
  __hip_bfloat16* y_bf  = (__hip_bfloat16*)alloc((size_t)MROWS * CC * 2);

  // prep: bf16 activations + transposed bf16 weights
  const size_t nx = (size_t)MROWS * CC;
  cvt_kernel<<<(unsigned)((nx + 255) / 256), 256, 0, stream>>>(x, x_bf, nx);
  transpose_cvt<<<(1024 * 128 + 255) / 256, 256, 0, stream>>>(query_w, qwT, 1024, 128);
  transpose_cvt<<<(1024 * 256 + 255) / 256, 256, 0, stream>>>(se1_w, se1T, 1024, 256);
  transpose_cvt<<<(256 * 1024 + 255) / 256, 256, 0, stream>>>(se2_w, se2T, 256, 1024);
  transpose_cvt<<<(1024 * 1024 + 255) / 256, 256, 0, stream>>>(out_w, owT, 1024, 1024);

  // stage 1: wave params + sample indices (tiny-N projection, VALU)
  wave_params_kernel<<<MROWS, 256, 0, stream>>>(x, wave_w, wave_b, freq, decay, sidx, vmask);

  // stage 2: queries = silu(x @ query_w + b)   [8192 x 128 x 1024]
  gemm_bf16_wmma<0, float><<<dim3(128 / 128, MROWS / 128), 256, 0, stream>>>(
      x_bf, qwT, query_b, nullptr, queries, MROWS, 128, 1024);

  // stage 3: adaptive gather-softmax conv -> out (f32 + bf16)
  attend_kernel<<<MROWS, 256, 0, stream>>>(x, queries, freq, decay, sidx, vmask, key_w, outf, outbf);

  // stage 4: t = silu(out @ se1_w + b)         [8192 x 256 x 1024]
  gemm_bf16_wmma<0, __hip_bfloat16><<<dim3(256 / 128, MROWS / 128), 256, 0, stream>>>(
      outbf, se1T, se1_b, nullptr, t_bf, MROWS, 256, 1024);

  // stage 5: y = sigmoid(t @ se2_w + b) * out  [8192 x 1024 x 256], fused SE multiply
  gemm_bf16_wmma<1, __hip_bfloat16><<<dim3(1024 / 128, MROWS / 128), 256, 0, stream>>>(
      t_bf, se2T, se2_b, outf, y_bf, MROWS, 1024, 256);

  // stage 6: final = silu(y @ out_w)           [8192 x 1024 x 1024]
  gemm_bf16_wmma<0, float><<<dim3(1024 / 128, MROWS / 128), 256, 0, stream>>>(
      y_bf, owT, nullptr, nullptr, (float*)d_out, MROWS, 1024, 1024);
}
